// EdgeTransition_30906584662428
// MI455X (gfx1250) — matrix operand order repaired and checked by reading, not compiled, not run on hardware
//
#include <hip/hip_runtime.h>
#include <hip/hip_bf16.h>

// ---------- plain clang ext-vector types (union-safe, no HIP ctors) ----------
typedef __bf16        bf16x16 __attribute__((ext_vector_type(16)));
typedef float         f32x8   __attribute__((ext_vector_type(8)));
typedef float         f32x4   __attribute__((ext_vector_type(4)));
typedef unsigned int  u32x4   __attribute__((ext_vector_type(4)));
typedef unsigned int  u32x2   __attribute__((ext_vector_type(2)));

#define N_SEQ     512
#define NODE_DIM  256
#define EDGE_IN   128
#define EDGE_OUT  128
#define BIAS_DIM  128
#define HIDDEN    384

#define TM        128          // rows (j's) per block
#define SH        392          // bf16 LDS stride for h/t (384 + 8 pad)
#define SO        132          // f32 LDS stride for out staging
#define SMEM_H    (TM * SH * 2)                 // 100352 B
#define SMEM_T    (TM * SH * 2)                 // 100352 B
#define SMEM_O    (TM * SO * 4)                 // 67584 B
#define SMEM_TOTAL (SMEM_H + SMEM_T + SMEM_O)   // 268288 B < 320 KB/WGP

__device__ __forceinline__ unsigned short f2bf(float f) {
  return __builtin_bit_cast(unsigned short, (__bf16)f);
}
__device__ __forceinline__ float bf2f(unsigned short u) {
  unsigned int x = ((unsigned int)u) << 16;
  return __builtin_bit_cast(float, x);
}

// Load a 32-byte WMMA fragment from two 16-byte chunks.
__device__ __forceinline__ bf16x16 ld_frag(const unsigned short* p0,
                                           const unsigned short* p1) {
  union { u32x4 q[2]; bf16x16 v; } u;
  u.q[0] = *(const u32x4*)p0;
  u.q[1] = *(const u32x4*)p1;
  return u.v;
}

#define WMMA_BF16(A, B, C) \
  __builtin_amdgcn_wmma_f32_16x16x32_bf16(false, (A), false, (B), (short)0, (C), false, false)

// ---------------- prep kernel: nb = node @ W_init + b (bf16) ----------------
__global__ void k_nb(const float* __restrict__ node,
                     const float* __restrict__ w_init,
                     const float* __restrict__ b_init,
                     unsigned short* __restrict__ nbbf) {
  const int r = blockIdx.x;      // 0..511
  const int c = threadIdx.x;     // 0..127
  float acc = b_init[c];
  for (int k = 0; k < NODE_DIM; ++k)
    acc = fmaf(node[r * NODE_DIM + k], w_init[k * BIAS_DIM + c], acc);
  nbbf[r * BIAS_DIM + c] = f2bf(acc);
}

// -------- prep kernel: wT[n][k] = bf16(w[k][n])  (K x Nn -> Nn x K) ---------
__global__ void k_tr(const float* __restrict__ w, unsigned short* __restrict__ wT,
                     int K, int Nn) {
  int idx = blockIdx.x * 256 + threadIdx.x;
  if (idx >= K * Nn) return;
  int n = idx / K, k = idx - n * K;
  wT[idx] = f2bf(w[(size_t)k * Nn + n]);
}

// ------------------------------- main kernel --------------------------------
// 8 waves arranged as 4(M) x 2(N): each wave owns 32 rows x half the output
// columns, so every B fragment feeds TWO wmma ops (halves B operand traffic).
__global__ __launch_bounds__(256, 1)
void edge_main(const float* __restrict__ edge,
               const unsigned short* __restrict__ nbbf,
               const unsigned short* __restrict__ wt0T,
               const unsigned short* __restrict__ wt1T,
               const unsigned short* __restrict__ wfinT,
               const float* __restrict__ b_t0,
               const float* __restrict__ b_t1,
               const float* __restrict__ b_fin,
               const float* __restrict__ ln_g,
               const float* __restrict__ ln_b,
               float* __restrict__ out) {
  extern __shared__ char smem[];
  unsigned short* sh = (unsigned short*)smem;              // h  : TM x SH bf16
  unsigned short* st = (unsigned short*)(smem + SMEM_H);   // t  : TM x SH bf16
  float*          so = (float*)(smem + SMEM_H + SMEM_T);   // out: TM x SO f32

  const int i   = blockIdx.y;
  const int j0  = blockIdx.x * TM;
  const int tid = threadIdx.x;

  // ---- build h = [edge_ij | nb_i | nb_j] in LDS (bf16) ----
  for (int v = tid; v < TM * 32; v += 256) {
    int r  = v >> 5;
    int c4 = (v & 31) << 2;
    f32x4 e = *(const f32x4*)&edge[((size_t)(i * N_SEQ + j0 + r)) * EDGE_IN + c4];
    u32x2 p;
    p.x = (unsigned int)f2bf(e.x) | ((unsigned int)f2bf(e.y) << 16);
    p.y = (unsigned int)f2bf(e.z) | ((unsigned int)f2bf(e.w) << 16);
    *(u32x2*)&sh[r * SH + c4] = p;
  }
  for (int v = tid; v < TM * 16; v += 256) {
    int r  = v >> 4;
    int c8 = (v & 15) << 3;
    *(u32x4*)&sh[r * SH + EDGE_IN + c8] = *(const u32x4*)&nbbf[i * BIAS_DIM + c8];
  }
  for (int v = tid; v < TM * 16; v += 256) {
    int r  = v >> 4;
    int c8 = (v & 15) << 3;
    *(u32x4*)&sh[r * SH + EDGE_IN + BIAS_DIM + c8] =
        *(const u32x4*)&nbbf[(j0 + r) * BIAS_DIM + c8];
  }
  __syncthreads();   // (1) h complete

  const int wave = tid >> 5;
  const int lane = tid & 31;
  const int ln15 = lane & 15;
  const int lhi  = lane >> 4;
  const int mq   = wave & 3;        // M quadrant: rows [32*mq, 32*mq+32)
  const int nh   = wave >> 2;       // N half
  const int rowA = mq * 32;

  bf16x16 a0[12], a1[12];

  // ===== GEMM1: t = relu(h @ w_t0 + b_t0) =====
  {
    const unsigned short* ar0 = &sh[(rowA + ln15) * SH + lhi * 8];
    const unsigned short* ar1 = ar0 + 16 * SH;
#pragma unroll
    for (int kt = 0; kt < 12; ++kt) {
      a0[kt] = ld_frag(ar0 + kt * 32, ar0 + kt * 32 + 16);
      a1[kt] = ld_frag(ar1 + kt * 32, ar1 + kt * 32 + 16);
    }
#pragma unroll 1
    for (int nt = nh * 12; nt < nh * 12 + 12; ++nt) {
      const unsigned short* brow = wt0T + (nt * 16 + ln15) * HIDDEN + lhi * 8;
      __builtin_prefetch(brow + 16 * HIDDEN, 0, 1);   // next n-tile (global_prefetch)
      float bias = b_t0[nt * 16 + ln15];
      f32x8 acc0, acc1;
#pragma unroll
      for (int e = 0; e < 8; ++e) { acc0[e] = bias; acc1[e] = bias; }
#pragma unroll
      for (int kt = 0; kt < 12; ++kt) {
        bf16x16 b = ld_frag(brow + kt * 32, brow + kt * 32 + 16);
        acc0 = WMMA_BF16(a0[kt], b, acc0);
        acc1 = WMMA_BF16(a1[kt], b, acc1);
      }
#pragma unroll
      for (int e = 0; e < 8; ++e) {
        int m = e + lhi * 8;
        float x0 = acc0[e] > 0.f ? acc0[e] : 0.f;
        float x1 = acc1[e] > 0.f ? acc1[e] : 0.f;
        st[(rowA + m) * SH + nt * 16 + ln15]      = f2bf(x0);
        st[(rowA + 16 + m) * SH + nt * 16 + ln15] = f2bf(x1);
      }
    }
  }
  __syncthreads();   // (2) t1 complete

  // ===== GEMM2: t = relu(t @ w_t1 + b_t1) + h =====
  {
    const unsigned short* ar0 = &st[(rowA + ln15) * SH + lhi * 8];
    const unsigned short* ar1 = ar0 + 16 * SH;
#pragma unroll
    for (int kt = 0; kt < 12; ++kt) {
      a0[kt] = ld_frag(ar0 + kt * 32, ar0 + kt * 32 + 16);
      a1[kt] = ld_frag(ar1 + kt * 32, ar1 + kt * 32 + 16);
    }
  }
  __syncthreads();   // (3) all waves cached t1 before st is overwritten
  {
#pragma unroll 1
    for (int nt = nh * 12; nt < nh * 12 + 12; ++nt) {
      const unsigned short* brow = wt1T + (nt * 16 + ln15) * HIDDEN + lhi * 8;
      __builtin_prefetch(brow + 16 * HIDDEN, 0, 1);
      float bias = b_t1[nt * 16 + ln15];
      f32x8 acc0, acc1;
#pragma unroll
      for (int e = 0; e < 8; ++e) { acc0[e] = bias; acc1[e] = bias; }
#pragma unroll
      for (int kt = 0; kt < 12; ++kt) {
        bf16x16 b = ld_frag(brow + kt * 32, brow + kt * 32 + 16);
        acc0 = WMMA_BF16(a0[kt], b, acc0);
        acc1 = WMMA_BF16(a1[kt], b, acc1);
      }
#pragma unroll
      for (int e = 0; e < 8; ++e) {
        int m = e + lhi * 8;
        float x0 = acc0[e] > 0.f ? acc0[e] : 0.f;
        float x1 = acc1[e] > 0.f ? acc1[e] : 0.f;
        x0 += bf2f(sh[(rowA + m) * SH + nt * 16 + ln15]);        // + h residual
        x1 += bf2f(sh[(rowA + 16 + m) * SH + nt * 16 + ln15]);
        st[(rowA + m) * SH + nt * 16 + ln15]      = f2bf(x0);
        st[(rowA + 16 + m) * SH + nt * 16 + ln15] = f2bf(x1);
      }
    }
  }
  __syncthreads();   // (4) t2+h complete

  // ===== GEMM3: out = (t+h) @ w_fin + b_fin -> f32 LDS staging =====
  {
    const unsigned short* ar0 = &st[(rowA + ln15) * SH + lhi * 8];
    const unsigned short* ar1 = ar0 + 16 * SH;
#pragma unroll
    for (int kt = 0; kt < 12; ++kt) {
      a0[kt] = ld_frag(ar0 + kt * 32, ar0 + kt * 32 + 16);
      a1[kt] = ld_frag(ar1 + kt * 32, ar1 + kt * 32 + 16);
    }
#pragma unroll 1
    for (int nt = nh * 4; nt < nh * 4 + 4; ++nt) {
      const unsigned short* brow = wfinT + (nt * 16 + ln15) * HIDDEN + lhi * 8;
      float bias = b_fin[nt * 16 + ln15];
      f32x8 acc0, acc1;
#pragma unroll
      for (int e = 0; e < 8; ++e) { acc0[e] = bias; acc1[e] = bias; }
#pragma unroll
      for (int kt = 0; kt < 12; ++kt) {
        bf16x16 b = ld_frag(brow + kt * 32, brow + kt * 32 + 16);
        acc0 = WMMA_BF16(a0[kt], b, acc0);
        acc1 = WMMA_BF16(a1[kt], b, acc1);
      }
#pragma unroll
      for (int e = 0; e < 8; ++e) {
        int m = e + lhi * 8;
        so[(rowA + m) * SO + nt * 16 + ln15]      = acc0[e];
        so[(rowA + 16 + m) * SO + nt * 16 + ln15] = acc1[e];
      }
    }
  }
  __syncthreads();   // (5) f32 out staging complete

  // ===== LayerNorm over 128 outputs, 2 adjacent lanes per row =====
  {
    const int r    = tid >> 1;     // 0..127 (pair stays inside one wave)
    const int half = lane & 1;
    const float* rowp = &so[r * SO + half * 64];
    float s = 0.f, q = 0.f;
#pragma unroll
    for (int c = 0; c < 64; c += 4) {
      f32x4 x = *(const f32x4*)&rowp[c];
      s += x.x + x.y + x.z + x.w;
      q += x.x * x.x + x.y * x.y + x.z * x.z + x.w * x.w;
    }
    s += __shfl_xor(s, 1);
    q += __shfl_xor(q, 1);
    const float mu   = s * (1.f / 128.f);
    const float var  = q * (1.f / 128.f) - mu * mu;
    const float rstd = rsqrtf(var + 1e-5f);

    float* op = &out[((size_t)(i * N_SEQ) + j0 + r) * EDGE_OUT + half * 64];
#pragma unroll
    for (int c = 0; c < 64; c += 4) {
      f32x4 x  = *(const f32x4*)&rowp[c];
      f32x4 g  = *(const f32x4*)&ln_g[half * 64 + c];
      f32x4 bb = *(const f32x4*)&ln_b[half * 64 + c];
      f32x4 y;
      y.x = (x.x - mu) * rstd * g.x + bb.x;
      y.y = (x.y - mu) * rstd * g.y + bb.y;
      y.z = (x.z - mu) * rstd * g.z + bb.z;
      y.w = (x.w - mu) * rstd * g.w + bb.w;
      *(f32x4*)&op[c] = y;
    }
  }
}

// -------------------------------- launcher ----------------------------------
extern "C" void kernel_launch(void* const* d_in, const int* in_sizes, int n_in,
                              void* d_out, int out_size, void* d_ws, size_t ws_size,
                              hipStream_t stream) {
  const float* node   = (const float*)d_in[0];
  const float* edge   = (const float*)d_in[1];
  const float* w_init = (const float*)d_in[2];
  const float* b_init = (const float*)d_in[3];
  const float* w_t0   = (const float*)d_in[4];
  const float* b_t0   = (const float*)d_in[5];
  const float* w_t1   = (const float*)d_in[6];
  const float* b_t1   = (const float*)d_in[7];
  const float* w_fin  = (const float*)d_in[8];
  const float* b_fin  = (const float*)d_in[9];
  const float* ln_g   = (const float*)d_in[10];
  const float* ln_b   = (const float*)d_in[11];
  float* out = (float*)d_out;

  // workspace carve (bf16 buffers, all 16B aligned)
  unsigned short* nbbf  = (unsigned short*)d_ws;               // 512*128
  unsigned short* wt0T  = nbbf + 512 * BIAS_DIM;               // 384*384
  unsigned short* wt1T  = wt0T + HIDDEN * HIDDEN;              // 384*384
  unsigned short* wfinT = wt1T + HIDDEN * HIDDEN;              // 128*384

  k_nb<<<N_SEQ, BIAS_DIM, 0, stream>>>(node, w_init, b_init, nbbf);
  k_tr<<<(HIDDEN * HIDDEN + 255) / 256, 256, 0, stream>>>(w_t0, wt0T, HIDDEN, HIDDEN);
  k_tr<<<(HIDDEN * HIDDEN + 255) / 256, 256, 0, stream>>>(w_t1, wt1T, HIDDEN, HIDDEN);
  k_tr<<<(HIDDEN * EDGE_OUT + 255) / 256, 256, 0, stream>>>(w_fin, wfinT, HIDDEN, EDGE_OUT);

  (void)hipFuncSetAttribute((const void*)edge_main,
                            hipFuncAttributeMaxDynamicSharedMemorySize, SMEM_TOTAL);
  dim3 grid(N_SEQ / TM, N_SEQ, 1);   // (4, 512)
  edge_main<<<grid, 256, SMEM_TOTAL, stream>>>(edge, nbbf, wt0T, wt1T, wfinT,
                                               b_t0, b_t1, b_fin, ln_g, ln_b, out);
}
